// StructuralContrastiveLoss_42442866819240
// MI455X (gfx1250) — matrix operand order repaired and checked by reading, not compiled, not run on hardware
//
#include <hip/hip_runtime.h>
#include <hip/hip_bf16.h>

#define BATCH 4096
#define DIM   512
#define SB    8192          // stacked rows = 2*BATCH
#define NT    64            // tile grid: SB/128
// TEMP = 0.5 -> scale = 2.0f applied post-GEMM

typedef __bf16           v16bf __attribute__((ext_vector_type(16)));
typedef float            v8f   __attribute__((ext_vector_type(8)));
typedef unsigned short   v8u16 __attribute__((ext_vector_type(8)));

union BOp {
    v16bf v;
    v8u16 h[2];
};

__device__ __forceinline__ unsigned short f32_to_bf16_rne(float f) {
    unsigned int u = __float_as_uint(f);
    u += 0x7FFFu + ((u >> 16) & 1u);   // round to nearest even (inputs are finite normals)
    return (unsigned short)(u >> 16);
}

// Async copy 16B global -> LDS (ASYNCcnt-tracked), GVS form: scalar base + 32-bit voffset.
__device__ __forceinline__ void async_cp16(unsigned lds_byte_addr, unsigned voff_bytes,
                                           const void* sbase) {
    asm volatile("global_load_async_to_lds_b128 %0, %1, %2"
                 :: "v"(lds_byte_addr), "v"(voff_bytes), "s"(sbase)
                 : "memory");
}
__device__ __forceinline__ void wait_async0() {
    asm volatile("s_wait_asynccnt 0x0" ::: "memory");
}

// ---------------------------------------------------------------- init: zero accumulators
__global__ void scl_init_kernel(float* red) {
    int i = blockIdx.x * blockDim.x + threadIdx.x;
    if (i < SB + BATCH + 1) red[i] = 0.0f;   // nfrow[8192], cnt[4096], S2
}

// ---------------------------------------------------------------- f32 -> bf16 stacked pack
__global__ void scl_convert_kernel(const float* __restrict__ zi,
                                   const float* __restrict__ zj,
                                   unsigned short* __restrict__ W) {
    int i = blockIdx.x * blockDim.x + threadIdx.x;   // grid covers BATCH*DIM exactly
    W[i]                       = f32_to_bf16_rne(zi[i]);
    W[(size_t)BATCH * DIM + i] = f32_to_bf16_rne(zj[i]);
}

// ---------------------------------------------------------------- cnt[r] = row sums of (Q==1)
__global__ __launch_bounds__(256) void scl_cnt_kernel(const int* __restrict__ Q,
                                                      float* __restrict__ cnt) {
    __shared__ float red[256];
    const int r = blockIdx.x;
    const int4* row = (const int4*)(Q + (size_t)r * BATCH);
    int c = 0;
    for (int idx = threadIdx.x; idx < BATCH / 4; idx += 256) {
        int4 qv = row[idx];
        c += (qv.x == 1) + (qv.y == 1) + (qv.z == 1) + (qv.w == 1);
    }
    red[threadIdx.x] = (float)c;
    __syncthreads();
    for (int s = 128; s > 0; s >>= 1) {
        if (threadIdx.x < s) red[threadIdx.x] += red[threadIdx.x + s];
        __syncthreads();
    }
    if (threadIdx.x == 0) cnt[r] = red[0];
}

// ---------------------------------------------------------------- symmetric (SYRK-style) main
// Upper-triangular tile pass over S = 2*W*W^T (8192x8192). WG = 128 threads = 4 waves (2x2),
// WG tile 128x128, wave tile 64x64 = 4x4 subtiles x 8 VGPRs = 128 accum VGPRs.
// Off-diagonal tiles mirror: column exp-sums -> nfrow[C], pos[cm,rm]*sval -> S2.
__global__ __launch_bounds__(128) void scl_main_kernel(
        const unsigned short* __restrict__ W,
        const int* __restrict__ Q,
        float* __restrict__ nfrow, float* __restrict__ S2) {

    // per buffer (shorts): A[128*32]=4096 (8KB), B[128*32]=4096 -> 16 KB; double buffered
    __shared__ unsigned short lds[2][8192];

    // ---- decode linear block index -> (ti, tj), ti <= tj ----
    int ti = 0, rem = blockIdx.x;
    while (rem >= NT - ti) { rem -= NT - ti; ++ti; }
    const int tj = ti + rem;
    const bool mirror = (ti != tj);
    const int M0 = ti * 128;
    const int N0 = tj * 128;

    const int t      = threadIdx.x;
    const int wave   = t >> 5;
    const int lane   = t & 31;
    const int lane16 = lane & 15;
    const int half   = lane >> 4;        // K-half selector per ISA operand layout
    const int wm     = wave >> 1;        // 0..1 (M sub: 64 rows)
    const int wn     = wave & 1;         // 0..1 (N sub: 64 cols)

    const unsigned ldsbase = (unsigned)(uintptr_t)(&lds[0][0]);

    v8f acc[4][4] = {};

    // ---- stage loads: 1024 16B chunks (A:512, B:512) over 128 threads = 8 async/thread ----
    auto stage_loads = [&](int k0, unsigned bufbase) {
        const unsigned short* sb = W + k0;   // uniform scalar base, advances per stage
#pragma unroll
        for (int s = 0; s < 4; ++s) {
            const int c   = t + 128 * s;     // chunk id 0..511
            const int row = c >> 2;          // 0..127
            const int cc  = (c & 3) * 8;     // element offset in 32-wide K chunk
            async_cp16(bufbase + (unsigned)c * 16u,
                       (unsigned)(((M0 + row) * DIM + cc) * 2), sb);
            async_cp16(bufbase + 8192u + (unsigned)c * 16u,
                       (unsigned)(((N0 + row) * DIM + cc) * 2), sb);
        }
    };

    stage_loads(0, ldsbase);   // prologue

    const int NSTAGES = DIM / 32;   // 16
    for (int ks = 0; ks < NSTAGES; ++ks) {
        wait_async0();        // my stage-ks loads landed
        __syncthreads();      // everyone's landed; prior-stage LDS reads done

        if (ks + 1 < NSTAGES)
            stage_loads((ks + 1) * 32, ldsbase + (unsigned)((ks + 1) & 1) * 16384u);

        unsigned short* Alds = &lds[ks & 1][0];
        unsigned short* Blds = Alds + 4096;

        BOp aop[4];
#pragma unroll
        for (int p = 0; p < 4; ++p) {
            const v8u16* rp = (const v8u16*)(Alds + (wm * 64 + p * 16 + lane16) * 32);
            aop[p].h[0] = rp[half];
            aop[p].h[1] = rp[2 + half];
        }
#pragma unroll
        for (int q = 0; q < 4; ++q) {
            BOp bop;
            const v8u16* cp = (const v8u16*)(Blds + (wn * 64 + q * 16 + lane16) * 32);
            bop.h[0] = cp[half];
            bop.h[1] = cp[2 + half];
#pragma unroll
            for (int p = 0; p < 4; ++p)
                acc[p][q] = __builtin_amdgcn_wmma_f32_16x16x32_bf16(false, aop[p].v, false, bop.v,
                                                                    (short)0, acc[p][q], false, false);
        }
    }

    // ---------------- epilogue ----------------
    float rowp[4][8];    // per (p, v): row exp-sums
    float colp[4];       // per q: column exp-sums (mirror only)
#pragma unroll
    for (int p = 0; p < 4; ++p)
#pragma unroll
        for (int v = 0; v < 8; ++v) rowp[p][v] = 0.f;
#pragma unroll
    for (int q = 0; q < 4; ++q) colp[q] = 0.f;
    float s2loc = 0.f;

#pragma unroll
    for (int p = 0; p < 4; ++p) {
        const int Rb  = M0 + wm * 64 + p * 16 + half * 8;   // + v
        const int rmb = Rb & (BATCH - 1);
#pragma unroll
        for (int q = 0; q < 4; ++q) {
            const int C  = N0 + wn * 64 + q * 16 + lane16;
            const int cm = C & (BATCH - 1);
            int4 qm0, qm1;                        // mirror row: Q[cm, rmb..rmb+7] (aligned)
            if (mirror) {
                qm0 = *(const int4*)(Q + (size_t)cm * BATCH + rmb);
                qm1 = *(const int4*)(Q + (size_t)cm * BATCH + rmb + 4);
            }
#pragma unroll
            for (int v = 0; v < 8; ++v) {
                const float sval = 2.0f * acc[p][q][v];
                const float e = __expf(sval);
                rowp[p][v] += e;
                colp[q]    += e;
                if (Q[(size_t)(rmb + v) * BATCH + cm] == 1) s2loc += sval;
                if (mirror) {
                    const int qv = (v < 4) ? ((const int*)&qm0)[v] : ((const int*)&qm1)[v - 4];
                    if (qv == 1) s2loc += sval;
                }
            }
        }
    }

    // row sums: butterfly within each 16-lane half
#pragma unroll
    for (int mask = 1; mask < 16; mask <<= 1)
#pragma unroll
        for (int p = 0; p < 4; ++p)
#pragma unroll
            for (int v = 0; v < 8; ++v)
                rowp[p][v] += __shfl_xor(rowp[p][v], mask, 32);
    // column sums: combine the two halves (each lane then holds full column sum)
#pragma unroll
    for (int q = 0; q < 4; ++q) colp[q] += __shfl_xor(colp[q], 16, 32);
    // scalar S2: full-wave butterfly
#pragma unroll
    for (int mask = 1; mask < 32; mask <<= 1)
        s2loc += __shfl_xor(s2loc, mask, 32);

    if (lane16 == 0) {
#pragma unroll
        for (int p = 0; p < 4; ++p) {
            const int rb = M0 + wm * 64 + p * 16 + half * 8;
#pragma unroll
            for (int v = 0; v < 8; ++v)
                atomicAdd(&nfrow[rb + v], rowp[p][v]);
        }
    }
    if (mirror && half == 0) {
#pragma unroll
        for (int q = 0; q < 4; ++q)
            atomicAdd(&nfrow[N0 + wn * 64 + q * 16 + lane16], colp[q]);
    }
    if (lane == 0) atomicAdd(S2, s2loc);
}

// ---------------------------------------------------------------- final scalar reduction
__global__ void scl_final_kernel(const float* __restrict__ nfrow,
                                 const float* __restrict__ cnt,
                                 const float* __restrict__ S2,
                                 float* __restrict__ out) {
    __shared__ float red[256];
    float p = 0.f;
    for (int r = threadIdx.x; r < BATCH; r += 256)
        p += 2.0f * (logf(nfrow[r]) + logf(nfrow[r + BATCH])) * cnt[r];
    red[threadIdx.x] = p;
    __syncthreads();
    for (int s = 128; s > 0; s >>= 1) {
        if (threadIdx.x < s) red[threadIdx.x] += red[threadIdx.x + s];
        __syncthreads();
    }
    if (threadIdx.x == 0) out[0] = (red[0] - S2[0]) / (float)(2 * BATCH);
}

// ---------------------------------------------------------------- launch
extern "C" void kernel_launch(void* const* d_in, const int* in_sizes, int n_in,
                              void* d_out, int out_size, void* d_ws, size_t ws_size,
                              hipStream_t stream) {
    const float* zi = (const float*)d_in[0];
    const float* zj = (const float*)d_in[1];
    const int*   Q  = (const int*)d_in[2];
    float*       out = (float*)d_out;

    char* ws = (char*)d_ws;
    unsigned short* W = (unsigned short*)ws;                       // stacked bf16, 8 MB
    float* red   = (float*)(ws + (size_t)SB * DIM * 2);
    float* nfrow = red;                 // [8192]
    float* cnt   = red + SB;            // [4096]
    float* S2    = red + SB + BATCH;    // [1]

    scl_init_kernel<<<(SB + BATCH + 1 + 255) / 256, 256, 0, stream>>>(red);
    scl_convert_kernel<<<(BATCH * DIM) / 256, 256, 0, stream>>>(zi, zj, W);
    scl_cnt_kernel<<<BATCH, 256, 0, stream>>>(Q, cnt);

    const int nblocks = NT * (NT + 1) / 2;   // 2080 upper-triangular 128x128 tiles
    scl_main_kernel<<<nblocks, 128, 0, stream>>>(W, Q, nfrow, S2);

    scl_final_kernel<<<1, 256, 0, stream>>>(nfrow, cnt, S2, out);
}
